// BiLSTMCRF_67250597920850
// MI455X (gfx1250) — compile-verified, matrix operand order checked by reading, not run on hardware
//
#include <hip/hip_runtime.h>
#include <hip/hip_bf16.h>
#include <math.h>

// Problem constants (match reference)
#define S_LEN  4096
#define EDIM   512
#define CEDIM  128
#define CHDIM  128
#define HDIM   512
#define NTAG   50
#define NTAGP  64          // padded tag count (multiple of 16)

typedef __attribute__((ext_vector_type(16))) _Float16 v16h;
typedef __attribute__((ext_vector_type(8)))  float    v8f;

__device__ __forceinline__ float sig_(float x) { return 1.0f / (1.0f + __expf(-x)); }

// ---- gfx1250 async copy helpers (inline asm; VFLAT async encoding) --------
// vdst = VGPR with LDS byte offset, vaddr = 64-bit global address, saddr = off
__device__ __forceinline__ void async_g2l_b128(unsigned lds_addr, const void* gaddr)
{
    asm volatile("global_load_async_to_lds_b128 %0, %1, off"
                 :: "v"(lds_addr), "v"(gaddr) : "memory");
}
__device__ __forceinline__ void wait_async_le2() {
    asm volatile("s_wait_asynccnt 0x2" ::: "memory");
}
__device__ __forceinline__ void wait_async_0() {
    asm volatile("s_wait_asynccnt 0x0" ::: "memory");
}

// ---------------------------------------------------------------------------
// Embedding gather -> f16 char input
__global__ void gather_char_k(const int* __restrict__ cs,
                              const float* __restrict__ char_emb,
                              _Float16* __restrict__ ce_h)
{
    int i = blockIdx.x * blockDim.x + threadIdx.x;          // S*CE
    if (i >= S_LEN * CEDIM) return;
    int t = i / CEDIM, k = i % CEDIM;
    ce_h[i] = (_Float16)char_emb[(size_t)cs[t] * CEDIM + k];
}

// Generic f32 -> f16 weight conversion
__global__ void f2h_k(const float* __restrict__ src, _Float16* __restrict__ dst, int n)
{
    int i = blockIdx.x * blockDim.x + threadIdx.x;
    if (i < n) dst[i] = (_Float16)src[i];
}

// W_tag [50,1024] -> zero-padded f16 [64,1024]
__global__ void wtag_pad_k(const float* __restrict__ W, _Float16* __restrict__ Wh)
{
    int i = blockIdx.x * blockDim.x + threadIdx.x;          // 64*1024
    if (i >= NTAGP * 2 * HDIM) return;
    int r = i >> 11, k = i & (2 * HDIM - 1);
    Wh[i] = (_Float16)((r < NTAG) ? W[(size_t)r * 2 * HDIM + k] : 0.0f);
}

// Build word-LSTM input  emb[t] = [ word_emb[sentence[t]] (512) | char_out[t] (256) ]  as f16
__global__ void build_emb_k(const int* __restrict__ sent,
                            const float* __restrict__ word_emb,
                            const _Float16* __restrict__ char_out_h,
                            _Float16* __restrict__ emb_h)
{
    int i = blockIdx.x * blockDim.x + threadIdx.x;          // S*768
    if (i >= S_LEN * 768) return;
    int t = i / 768, k = i % 768;
    if (k < EDIM)
        emb_h[i] = (_Float16)word_emb[(size_t)sent[t] * EDIM + k];
    else
        emb_h[i] = char_out_h[(size_t)t * 256 + (k - EDIM)];
}

// ---------------------------------------------------------------------------
// WMMA GEMM:  C[M,Nvalid] = A[M,K] (f16 row-major) @ W[N,K]^T (f16 row-major) + bias
// 128-thread block (4 waves) computes a 64x64 tile: wave w owns rows w*16..w*16+15,
// all four 16-wide N sub-tiles.  The shared 64x32 B slab is double-buffered in LDS
// via global_load_async_to_lds_b128 (ASYNCcnt-synchronized).  K%32==0, M%64==0, N%64==0.
__global__ void wmma_gemm_xWT(const _Float16* __restrict__ A,
                              const _Float16* __restrict__ W,
                              const float* __restrict__ bias,
                              float* __restrict__ C,
                              int M, int N, int K, int Nvalid)
{
    __shared__ _Float16 bbuf[2][64 * 32];        // 2 x 4KB slabs

    const int tid  = threadIdx.x;                // 128 threads
    const int lane = tid & 31;
    const int wave = tid >> 5;
    const int ml   = lane & 15;
    const int hi   = lane >> 4;

    const int col0 = blockIdx.x * 64;
    const int row0 = blockIdx.y * 64 + wave * 16;

    const _Float16* Ap = A + (size_t)(row0 + ml) * K;

    // each thread stages two 16B chunks of the 4KB slab (chunks tid and tid+128)
    const int n0 = tid >> 2,         o0 = (tid & 3) * 8;          // 8 halfs = 16B
    const int n1 = (tid + 128) >> 2, o1 = ((tid + 128) & 3) * 8;
    const unsigned lbase0 = (unsigned)(size_t)(&bbuf[0][0]);
    const unsigned lbase1 = (unsigned)(size_t)(&bbuf[1][0]);

    v8f acc0 = {}, acc1 = {}, acc2 = {}, acc3 = {};

    const int nk = K / 32;
    {   // prologue: stage slab 0
        const _Float16* g0 = W + (size_t)(col0 + n0) * K + o0;
        const _Float16* g1 = W + (size_t)(col0 + n1) * K + o1;
        async_g2l_b128(lbase0 + (unsigned)(n0 * 32 + o0) * 2u, g0);
        async_g2l_b128(lbase0 + (unsigned)(n1 * 32 + o1) * 2u, g1);
    }

    for (int i = 0; i < nk; ++i) {
        __syncthreads();                         // prior compute done before overwrite
        if (i + 1 < nk) {
            const int k0 = (i + 1) * 32;
            const unsigned lb = (i + 1) & 1 ? lbase1 : lbase0;
            const _Float16* g0 = W + (size_t)(col0 + n0) * K + k0 + o0;
            const _Float16* g1 = W + (size_t)(col0 + n1) * K + k0 + o1;
            async_g2l_b128(lb + (unsigned)(n0 * 32 + o0) * 2u, g0);
            async_g2l_b128(lb + (unsigned)(n1 * 32 + o1) * 2u, g1);
            wait_async_le2();                    // slab i complete (in-order), slab i+1 in flight
        } else {
            wait_async_0();
        }
        __syncthreads();                         // slab i visible to all waves

        const _Float16* lb = &bbuf[i & 1][0];
        const _Float16* ap = Ap + i * 32;
        v16h a, b0, b1, b2, b3;
#pragma unroll
        for (int e = 0; e < 16; ++e) {
            const int k = e + ((e < 8) ? 0 : 8) + hi * 8;   // ISA 16-bit A 16x32 layout
            a[e]  = ap[k];
            b0[e] = lb[(0 * 16 + ml) * 32 + k];
            b1[e] = lb[(1 * 16 + ml) * 32 + k];
            b2[e] = lb[(2 * 16 + ml) * 32 + k];
            b3[e] = lb[(3 * 16 + ml) * 32 + k];
        }
        acc0 = __builtin_amdgcn_wmma_f32_16x16x32_f16(false, a, false, b0, (short)0, acc0, false, false);
        acc1 = __builtin_amdgcn_wmma_f32_16x16x32_f16(false, a, false, b1, (short)0, acc1, false, false);
        acc2 = __builtin_amdgcn_wmma_f32_16x16x32_f16(false, a, false, b2, (short)0, acc2, false, false);
        acc3 = __builtin_amdgcn_wmma_f32_16x16x32_f16(false, a, false, b3, (short)0, acc3, false, false);
    }

    // store (C/D 16x16 f32 layout: n = lane&15, m = r + (lane>>4)*8)
    {
        const int nn[4] = { col0 + ml, col0 + 16 + ml, col0 + 32 + ml, col0 + 48 + ml };
        v8f accs[4] = { acc0, acc1, acc2, acc3 };
#pragma unroll
        for (int s = 0; s < 4; ++s) {
            const int n = nn[s];
            if (n < Nvalid) {
                const float bv = bias[n];
#pragma unroll
                for (int r = 0; r < 8; ++r)
                    C[(size_t)(row0 + r + hi * 8) * Nvalid + n] = accs[s][r] + bv;
            }
        }
    }
}

// ---------------------------------------------------------------------------
// Char BiLSTM recurrence.  One 512-thread block per direction (blockIdx.x).
__global__ void char_lstm_k(const float* __restrict__ gx,
                            const float* __restrict__ Whh_f,
                            const float* __restrict__ Whh_b,
                            const float* __restrict__ bias_f,
                            const float* __restrict__ bias_b,
                            _Float16* __restrict__ char_out_h)
{
    const int dir = blockIdx.x;
    const float* gxd  = gx + (size_t)dir * S_LEN * 4 * CHDIM;
    const float* Whh  = dir ? Whh_b : Whh_f;
    const float* bias = dir ? bias_b : bias_f;

    __shared__ float h[CHDIM], c[CHDIM], g[4 * CHDIM];
    const int j = threadIdx.x;                   // 512 threads
    if (j < CHDIM) { h[j] = 0.0f; c[j] = 0.0f; }
    __syncthreads();

    const float4* wr = (const float4*)(Whh + (size_t)j * CHDIM);
    const float   bj = bias[j];

    for (int t = 0; t < S_LEN; ++t) {
        const int pos = dir ? (S_LEN - 1 - t) : t;
        if (t + 1 < S_LEN)
            __builtin_prefetch(&gxd[(size_t)(dir ? pos - 1 : pos + 1) * (4 * CHDIM) + j]);
        float acc = gxd[(size_t)pos * (4 * CHDIM) + j] + bj;
#pragma unroll 8
        for (int k4 = 0; k4 < CHDIM / 4; ++k4) {
            float4 w = wr[k4];
            acc += w.x * h[4 * k4 + 0] + w.y * h[4 * k4 + 1] +
                   w.z * h[4 * k4 + 2] + w.w * h[4 * k4 + 3];
        }
        g[j] = acc;
        __syncthreads();
        if (j < CHDIM) {
            float ig = sig_(g[j]);
            float fg = sig_(g[CHDIM + j]);
            float gg = tanhf(g[2 * CHDIM + j]);
            float og = sig_(g[3 * CHDIM + j]);
            float cn = fg * c[j] + ig * gg;
            c[j] = cn;
            float hn = og * tanhf(cn);
            h[j] = hn;
            char_out_h[(size_t)pos * 256 + dir * CHDIM + j] = (_Float16)hn;
        }
        __syncthreads();
    }
}

// Word BiLSTM recurrence.  One 1024-thread block per direction.
__global__ void word_lstm_k(const float* __restrict__ gx,
                            const float* __restrict__ Whh_f,
                            const float* __restrict__ Whh_b,
                            const float* __restrict__ bias_f,
                            const float* __restrict__ bias_b,
                            _Float16* __restrict__ lstm_out_h)
{
    const int dir = blockIdx.x;
    const float* gxd  = gx + (size_t)dir * S_LEN * 4 * HDIM;
    const float* Whh  = dir ? Whh_b : Whh_f;
    const float* bias = dir ? bias_b : bias_f;

    __shared__ float h[HDIM], c[HDIM], g[4 * HDIM];
    const int j  = threadIdx.x;                  // 1024 threads
    const int j2 = j + 1024;
    if (j < HDIM) { h[j] = 0.0f; c[j] = 0.0f; }
    __syncthreads();

    const float4* w0 = (const float4*)(Whh + (size_t)j  * HDIM);
    const float4* w1 = (const float4*)(Whh + (size_t)j2 * HDIM);
    const float   b0 = bias[j], b1 = bias[j2];

    for (int t = 0; t < S_LEN; ++t) {
        const int pos = dir ? (S_LEN - 1 - t) : t;
        if (t + 1 < S_LEN) {
            const float* nx = &gxd[(size_t)(dir ? pos - 1 : pos + 1) * (4 * HDIM)];
            __builtin_prefetch(nx + j);
            __builtin_prefetch(nx + j2);
        }
        float acc0 = gxd[(size_t)pos * (4 * HDIM) + j ] + b0;
        float acc1 = gxd[(size_t)pos * (4 * HDIM) + j2] + b1;
#pragma unroll 4
        for (int k4 = 0; k4 < HDIM / 4; ++k4) {
            float4 wa = w0[k4], wb = w1[k4];
            float h0 = h[4 * k4 + 0], h1 = h[4 * k4 + 1];
            float h2 = h[4 * k4 + 2], h3 = h[4 * k4 + 3];
            acc0 += wa.x * h0 + wa.y * h1 + wa.z * h2 + wa.w * h3;
            acc1 += wb.x * h0 + wb.y * h1 + wb.z * h2 + wb.w * h3;
        }
        g[j] = acc0;
        g[j2] = acc1;
        __syncthreads();
        if (j < HDIM) {
            float ig = sig_(g[j]);
            float fg = sig_(g[HDIM + j]);
            float gg = tanhf(g[2 * HDIM + j]);
            float og = sig_(g[3 * HDIM + j]);
            float cn = fg * c[j] + ig * gg;
            c[j] = cn;
            float hn = og * tanhf(cn);
            h[j] = hn;
            lstm_out_h[(size_t)pos * (2 * HDIM) + dir * HDIM + j] = (_Float16)hn;
        }
        __syncthreads();
    }
}

// ---------------------------------------------------------------------------
// CRF Viterbi: forward max-plus + backtrack.  Single 64-thread block.
__global__ void viterbi_k(const float* __restrict__ emis,     // [S][50]
                          const float* __restrict__ start_t,  // [50]
                          const float* __restrict__ end_t,    // [50]
                          const float* __restrict__ trans,    // [50][50]
                          int* __restrict__ bp,               // [S-1][50]
                          int* __restrict__ path)             // [S]
{
    __shared__ float transc[NTAG * 52];     // transposed, padded stride
    __shared__ float score[NTAGP], nscore[NTAGP];
    const int j = threadIdx.x;              // 64 threads

    for (int idx = j; idx < NTAG * NTAG; idx += 64) {
        int i = idx / NTAG, jj = idx % NTAG;
        transc[jj * 52 + i] = trans[idx];   // transc[cur][prev]
    }
    if (j < NTAG) score[j] = start_t[j] + emis[j];
    __syncthreads();

    for (int t = 1; t < S_LEN; ++t) {
        if (j < NTAG) {
            float best = -3.0e38f; int bi = 0;
            const float* tc = &transc[j * 52];
#pragma unroll 10
            for (int i = 0; i < NTAG; ++i) {
                float v = score[i] + tc[i];
                if (v > best) { best = v; bi = i; }
            }
            nscore[j] = best + emis[(size_t)t * NTAG + j];
            bp[(size_t)(t - 1) * NTAG + j] = bi;
        }
        __syncthreads();
        if (j < NTAG) score[j] = nscore[j];
        __syncthreads();
    }

    if (j == 0) {
        float best = -3.0e38f; int last = 0;
        for (int i = 0; i < NTAG; ++i) {
            float v = score[i] + end_t[i];
            if (v > best) { best = v; last = i; }
        }
        path[S_LEN - 1] = last;
        for (int t = S_LEN - 2; t >= 0; --t) {
            last = bp[(size_t)t * NTAG + last];
            path[t] = last;
        }
    }
}

// ---------------------------------------------------------------------------
extern "C" void kernel_launch(void* const* d_in, const int* in_sizes, int n_in,
                              void* d_out, int out_size, void* d_ws, size_t ws_size,
                              hipStream_t stream)
{
    // inputs (setup_inputs dict order)
    const int*   sentence  = (const int*)  d_in[0];
    const int*   char_sent = (const int*)  d_in[1];
    const float* word_emb  = (const float*)d_in[2];
    const float* char_emb  = (const float*)d_in[3];
    const float* cWih_f = (const float*)d_in[4];
    const float* cWhh_f = (const float*)d_in[5];
    const float* cb_f   = (const float*)d_in[6];
    const float* cWih_b = (const float*)d_in[7];
    const float* cWhh_b = (const float*)d_in[8];
    const float* cb_b   = (const float*)d_in[9];
    const float* wWih_f = (const float*)d_in[10];
    const float* wWhh_f = (const float*)d_in[11];
    const float* wb_f   = (const float*)d_in[12];
    const float* wWih_b = (const float*)d_in[13];
    const float* wWhh_b = (const float*)d_in[14];
    const float* wb_b   = (const float*)d_in[15];
    const float* W_tag  = (const float*)d_in[16];
    const float* b_tag  = (const float*)d_in[17];
    const float* start_t = (const float*)d_in[18];
    const float* end_t   = (const float*)d_in[19];
    const float* trans   = (const float*)d_in[20];
    int* path = (int*)d_out;
    (void)in_sizes; (void)n_in; (void)out_size; (void)ws_size;

    // workspace carve (all 256B aligned)
    char* wsp = (char*)d_ws;
    auto carve = [&](size_t bytes) -> void* {
        void* p = (void*)wsp;
        wsp += (bytes + 255) & ~(size_t)255;
        return p;
    };
    _Float16* ce_h       = (_Float16*)carve((size_t)S_LEN * CEDIM * 2);          // 1 MB
    _Float16* cWih_h     = (_Float16*)carve((size_t)2 * 4 * CHDIM * CEDIM * 2);  // 0.25 MB
    float*    cgx        = (float*)   carve((size_t)2 * S_LEN * 4 * CHDIM * 4);  // 16 MB
    _Float16* char_out_h = (_Float16*)carve((size_t)S_LEN * 256 * 2);            // 2 MB
    _Float16* emb_h      = (_Float16*)carve((size_t)S_LEN * 768 * 2);            // 6 MB
    _Float16* wWih_h     = (_Float16*)carve((size_t)2 * 4 * HDIM * 768 * 2);     // 6 MB
    float*    wgx        = (float*)   carve((size_t)2 * S_LEN * 4 * HDIM * 4);   // 64 MB
    _Float16* lstm_out_h = (_Float16*)carve((size_t)S_LEN * 2 * HDIM * 2);       // 8 MB
    _Float16* wtag_h     = (_Float16*)carve((size_t)NTAGP * 2 * HDIM * 2);       // 128 KB
    float*    emis       = (float*)   carve((size_t)S_LEN * NTAG * 4);           // 0.8 MB
    int*      bp         = (int*)     carve((size_t)(S_LEN - 1) * NTAG * 4);     // 0.8 MB

    const int TPB = 256;

    // 1) embedding gather + weight conversions
    gather_char_k<<<(S_LEN * CEDIM + TPB - 1) / TPB, TPB, 0, stream>>>(char_sent, char_emb, ce_h);
    {
        int n = 4 * CHDIM * CEDIM;     // 65536 per dir
        f2h_k<<<(n + TPB - 1) / TPB, TPB, 0, stream>>>(cWih_f, cWih_h, n);
        f2h_k<<<(n + TPB - 1) / TPB, TPB, 0, stream>>>(cWih_b, cWih_h + n, n);
    }
    {
        int n = 4 * HDIM * 768;        // 1,572,864 per dir
        f2h_k<<<(n + TPB - 1) / TPB, TPB, 0, stream>>>(wWih_f, wWih_h, n);
        f2h_k<<<(n + TPB - 1) / TPB, TPB, 0, stream>>>(wWih_b, wWih_h + n, n);
    }
    wtag_pad_k<<<(NTAGP * 2 * HDIM + TPB - 1) / TPB, TPB, 0, stream>>>(W_tag, wtag_h);

    // 2) char input GEMMs (WMMA + async LDS staging): [4096,128] x [128,512]^T per dir
    {
        dim3 grid(4 * CHDIM / 64, S_LEN / 64);
        wmma_gemm_xWT<<<grid, 128, 0, stream>>>(ce_h, cWih_h, cb_f,
                                                cgx, S_LEN, 4 * CHDIM, CEDIM, 4 * CHDIM);
        wmma_gemm_xWT<<<grid, 128, 0, stream>>>(ce_h, cWih_h + (size_t)4 * CHDIM * CEDIM, cb_b,
                                                cgx + (size_t)S_LEN * 4 * CHDIM,
                                                S_LEN, 4 * CHDIM, CEDIM, 4 * CHDIM);
    }

    // 3) char recurrence (2 persistent blocks, fwd + bwd)
    char_lstm_k<<<2, 4 * CHDIM, 0, stream>>>(cgx, cWhh_f, cWhh_b, cb_f, cb_b, char_out_h);

    // 4) word input build + GEMMs: [4096,768] x [768,2048]^T per dir
    build_emb_k<<<(S_LEN * 768 + TPB - 1) / TPB, TPB, 0, stream>>>(sentence, word_emb, char_out_h, emb_h);
    {
        dim3 grid(4 * HDIM / 64, S_LEN / 64);
        wmma_gemm_xWT<<<grid, 128, 0, stream>>>(emb_h, wWih_h, wb_f,
                                                wgx, S_LEN, 4 * HDIM, 768, 4 * HDIM);
        wmma_gemm_xWT<<<grid, 128, 0, stream>>>(emb_h, wWih_h + (size_t)4 * HDIM * 768, wb_b,
                                                wgx + (size_t)S_LEN * 4 * HDIM,
                                                S_LEN, 4 * HDIM, 768, 4 * HDIM);
    }

    // 5) word recurrence (2 persistent blocks)
    word_lstm_k<<<2, 1024, 0, stream>>>(wgx, wWhh_f, wWhh_b, wb_f, wb_b, lstm_out_h);

    // 6) emissions GEMM (N padded to 64, valid 50): [4096,1024] x [1024,50]^T
    {
        dim3 grid(1, S_LEN / 64);
        wmma_gemm_xWT<<<grid, 128, 0, stream>>>(lstm_out_h, wtag_h, b_tag,
                                                emis, S_LEN, NTAGP, 2 * HDIM, NTAG);
    }

    // 7) Viterbi decode
    viterbi_k<<<1, 64, 0, stream>>>(emis, start_t, end_t, trans, bp, path);
}
